// DifferentialAttention_17927193493796
// MI455X (gfx1250) — matrix operand order repaired and checked by reading, not compiled, not run on hardware
//
#include <hip/hip_runtime.h>
#include <cstdint>
#include <cmath>

// Problem constants (B,S,H)=(2,2048,2048), NH=16, HD=64
#define BB  2
#define SS  2048
#define HH  2048
#define NHH 16
#define HDD 64

typedef __attribute__((ext_vector_type(16))) __bf16          bf16x16;
typedef __attribute__((ext_vector_type(8)))  float           f32x8;
typedef __attribute__((ext_vector_type(8)))  unsigned short  u16x8;
typedef __attribute__((ext_vector_type(16))) unsigned short  u16x16;

__device__ __forceinline__ unsigned short f2bf(float f) {
  unsigned int u = __float_as_uint(f);
  u += 0x7FFFu + ((u >> 16) & 1u);   // round-to-nearest-even
  return (unsigned short)(u >> 16);
}

// ---- WMMA fragment loaders (CDNA5 16x16x32 bf16 layouts, wave32) ----------
// A (16x32 MxK): lane L holds row (L&15); lanes<16: K in {0..7,16..23},
// lanes>=16: K in {8..15,24..31}.  Two b128 loads per lane.
__device__ __forceinline__ bf16x16 load_frag_a(const unsigned short* p, int ld, int lane) {
  const int row = lane & 15;
  const int kh  = (lane >> 4) << 3;            // 0 or 8
  const unsigned short* r = p + (size_t)row * ld;
  u16x8 lo = *(const u16x8*)(r + kh);          // K = kh .. kh+7
  u16x8 hi = *(const u16x8*)(r + 16 + kh);     // K = 16+kh .. 23+kh
  u16x16 v = __builtin_shufflevector(lo, hi, 0,1,2,3,4,5,6,7,8,9,10,11,12,13,14,15);
  return __builtin_bit_cast(bf16x16, v);
}

// B (32x16 KxN) sourced from "transposed" storage T[n][k] (k contiguous):
// lane L holds col (L&15); lanes<16: K in {0..7,16..23}, lanes>=16: K in {8..15,24..31}.
__device__ __forceinline__ bf16x16 load_frag_bt(const unsigned short* p, int ld, int lane) {
  const int col = lane & 15;
  const int k0  = (lane >> 4) << 3;            // 0 or 8
  const unsigned short* r = p + (size_t)col * ld + k0;
  u16x8 lo = *(const u16x8*)(r);
  u16x8 hi = *(const u16x8*)(r + 16);
  u16x16 v = __builtin_shufflevector(lo, hi, 0,1,2,3,4,5,6,7,8,9,10,11,12,13,14,15);
  return __builtin_bit_cast(bf16x16, v);
}

__device__ __forceinline__ f32x8 wmma_bf16(bf16x16 a, bf16x16 b, f32x8 c) {
  return __builtin_amdgcn_wmma_f32_16x16x32_bf16(false, a, false, b, (short)0, c, false, false);
}

// ---- Stage 0: f32 -> bf16 casts -------------------------------------------
__global__ __launch_bounds__(256) void cast_bf16_kernel(const float* __restrict__ x,
                                                        unsigned short* __restrict__ y, int n) {
  int i = blockIdx.x * 256 + threadIdx.x;
  if (i < n) y[i] = f2bf(x[i]);
}

// W (K x N) f32 -> W^T (N x K) bf16, 32x32 tiles via LDS
__global__ __launch_bounds__(256) void cast_transpose_kernel(const float* __restrict__ w,
                                                             unsigned short* __restrict__ wt,
                                                             int K, int N) {
  __shared__ float tile[32][33];
  const int bx = blockIdx.x * 32;  // N
  const int by = blockIdx.y * 32;  // K
  const int tx = threadIdx.x, ty = threadIdx.y;
#pragma unroll
  for (int r = ty; r < 32; r += 8)
    tile[r][tx] = w[(size_t)(by + r) * N + bx + tx];
  __syncthreads();
#pragma unroll
  for (int r = ty; r < 32; r += 8)
    wt[(size_t)(bx + r) * K + by + tx] = f2bf(tile[tx][r]);
}

// ---- Stage 1: projection GEMMs  C = Xbf(4096x2048) @ W  -------------------
// Each wave produces a 32x64 tile: 2 A-frags x 4 B-frags = 8 WMMAs / k-step.
// Ping-pong (even/odd) buffering, unrolled over two k-steps: each buffer is
// refilled in place right after its WMMAs consume it -> no rotation copies,
// loads stay ~8 WMMAs ahead of their consuming wait.
// vmode 0: write (b, head, s, d)   [Q and K]
// vmode 1: write (b, head, d, s)   [V transposed for the PV GEMM]
__global__ __launch_bounds__(256) void proj_gemm_kernel(const unsigned short* __restrict__ A,
                                                        const unsigned short* __restrict__ WT,
                                                        unsigned short* __restrict__ dst,
                                                        int N, int vmode) {
  const int lane = threadIdx.x & 31;
  const int wid  = blockIdx.x * 8 + (threadIdx.x >> 5);
  const int row0 = (wid & 127) * 32;          // 4096/32 = 128 row tiles
  const int col0 = (wid >> 7) * 64;

  auto loadA = [&](bf16x16* a, int kk) {
#pragma unroll
    for (int i = 0; i < 2; ++i)
      a[i] = load_frag_a(A + (size_t)(row0 + i * 16) * HH + kk, HH, lane);
  };
  auto loadB = [&](bf16x16* b, int kk) {
#pragma unroll
    for (int j = 0; j < 4; ++j)
      b[j] = load_frag_bt(WT + (size_t)(col0 + j * 16) * HH + kk, HH, lane);
  };

  f32x8 acc[2][4] = {};
  bf16x16 aE[2], bE[4], aO[2], bO[4];
  loadA(aE, 0);
  loadB(bE, 0);
  loadA(aO, 32);
  loadB(bO, 32);
  for (int kk = 0; kk + 128 <= HH; kk += 64) {
#pragma unroll
    for (int i = 0; i < 2; ++i)
#pragma unroll
      for (int j = 0; j < 4; ++j)
        acc[i][j] = wmma_bf16(aE[i], bE[j], acc[i][j]);
    loadA(aE, kk + 64);                        // refill E while O computes
    loadB(bE, kk + 64);
#pragma unroll
    for (int i = 0; i < 2; ++i)
#pragma unroll
      for (int j = 0; j < 4; ++j)
        acc[i][j] = wmma_bf16(aO[i], bO[j], acc[i][j]);
    loadA(aO, kk + 96);                        // refill O while E computes
    loadB(bO, kk + 96);
  }
#pragma unroll
  for (int i = 0; i < 2; ++i)
#pragma unroll
    for (int j = 0; j < 4; ++j)
      acc[i][j] = wmma_bf16(aE[i], bE[j], acc[i][j]);
#pragma unroll
  for (int i = 0; i < 2; ++i)
#pragma unroll
    for (int j = 0; j < 4; ++j)
      acc[i][j] = wmma_bf16(aO[i], bO[j], acc[i][j]);

  const int m0 = (lane >> 4) * 8;
  const int b  = row0 >> 11;                  // tiles never straddle the batch
  const int nheads = N >> 6;
#pragma unroll
  for (int i = 0; i < 2; ++i) {
    const int s0 = ((row0 + i * 16) & 2047) + m0;
#pragma unroll
    for (int j = 0; j < 4; ++j) {
      const int col  = col0 + j * 16 + (lane & 15);
      const int head = col >> 6;
      const int d    = col & 63;
      if (vmode == 0) {
        unsigned short* o = dst + ((size_t)(b * nheads + head) * SS) * HDD + d;
#pragma unroll
        for (int v = 0; v < 8; ++v) o[(size_t)(s0 + v) * HDD] = f2bf(acc[i][j][v]);
      } else {
        u16x8 pk;
#pragma unroll
        for (int v = 0; v < 8; ++v) pk[v] = f2bf(acc[i][j][v]);
        *(u16x8*)(dst + ((size_t)(b * nheads + head) * HDD + d) * SS + s0) = pk;
      }
    }
  }
}

// ---- Stage 2: causal differential flash-attention -------------------------
// 4 waves / block, each wave owns one (b, h, 16-row q tile) and a private LDS
// slice (per-wave LDS is in-order, so no block barrier is needed).
// Per 32-key step: 8 grouped K-frag loads -> 8 score WMMAs; V-frag loads are
// hoisted before the softmax VALU work to overlap global latency with exp().
__global__ __launch_bounds__(128) void diff_attn_kernel(const unsigned short* __restrict__ Q,
                                                        const unsigned short* __restrict__ K,
                                                        const unsigned short* __restrict__ Vt,
                                                        const float* __restrict__ lambda_param,
                                                        unsigned short* __restrict__ AO) {
  __shared__ alignas(16) unsigned short pTall[4][16 * 32];
  const int lane = threadIdx.x & 31;
  const int wv   = threadIdx.x >> 5;
  unsigned short* pT = pTall[wv];
  const int q0 = (blockIdx.x * 4 + wv) * 16;
  const int h  = blockIdx.y;
  const int b  = blockIdx.z;

  const size_t qk1 = (size_t)(b * 32 + h)      * SS * HDD;   // head h       (q1/k1)
  const size_t qk2 = (size_t)(b * 32 + 16 + h) * SS * HDD;   // head h+NH    (q2/k2)
  const size_t vbh = (size_t)(b * 16 + h)      * HDD * SS;   // V^T (d, s)

  const float lam     = tanhf(log1pf(expf(lambda_param[0])));  // tanh(softplus(x))
  const float scaling = 0.125f;                                // HD^-0.5

  bf16x16 a1[2], a2[2];
#pragma unroll
  for (int kk = 0; kk < 2; ++kk) {
    a1[kk] = load_frag_a(Q + qk1 + (size_t)q0 * HDD + kk * 32, HDD, lane);
    a2[kk] = load_frag_a(Q + qk2 + (size_t)q0 * HDD + kk * 32, HDD, lane);
  }

  f32x8 acc[4] = {};
  float mrow[8], lrow[8];
#pragma unroll
  for (int v = 0; v < 8; ++v) { mrow[v] = -__builtin_inff(); lrow[v] = 0.0f; }

  const int h8  = (lane >> 4) * 8;   // row base this lane owns in C layout
  const int nlo = lane & 15;         // column this lane owns in C layout

  for (int kt = 0; kt < q0 + 16; kt += 32) {
    if (kt + 32 < q0 + 16) {         // hint next K tiles toward the caches
      __builtin_prefetch(K + qk1 + (size_t)(kt + 32) * HDD, 0, 1);
      __builtin_prefetch(K + qk2 + (size_t)(kt + 32) * HDD, 0, 1);
    }
    // grouped K-fragment loads: one wait, then 8 back-to-back WMMAs
    bf16x16 bk1[2][2], bk2[2][2];
#pragma unroll
    for (int t = 0; t < 2; ++t)
#pragma unroll
      for (int kk = 0; kk < 2; ++kk) {
        bk1[t][kk] = load_frag_bt(K + qk1 + (size_t)(kt + t * 16) * HDD + kk * 32, HDD, lane);
        bk2[t][kk] = load_frag_bt(K + qk2 + (size_t)(kt + t * 16) * HDD + kk * 32, HDD, lane);
      }
    f32x8 s1[2] = {}, s2[2] = {};
#pragma unroll
    for (int t = 0; t < 2; ++t)
#pragma unroll
      for (int kk = 0; kk < 2; ++kk) {
        s1[t] = wmma_bf16(a1[kk], bk1[t][kk], s1[t]);
        s2[t] = wmma_bf16(a2[kk], bk2[t][kk], s2[t]);
      }
    // V fragments issued early: global latency overlaps the softmax VALU work
    bf16x16 bv[4];
#pragma unroll
    for (int j = 0; j < 4; ++j)
      bv[j] = load_frag_bt(Vt + vbh + (size_t)(j * 16) * SS + kt, SS, lane);

#pragma unroll
    for (int v = 0; v < 8; ++v) {
      const int qrow = q0 + h8 + v;
      float a0 = (s1[0][v] - lam * s2[0][v]) * scaling;
      float a1v = (s1[1][v] - lam * s2[1][v]) * scaling;
      if (kt + nlo      > qrow) a0  = -__builtin_inff();   // causal mask
      if (kt + 16 + nlo > qrow) a1v = -__builtin_inff();
      float mx = fmaxf(a0, a1v);
      mx = fmaxf(mx, __shfl_xor(mx, 1, 32));
      mx = fmaxf(mx, __shfl_xor(mx, 2, 32));
      mx = fmaxf(mx, __shfl_xor(mx, 4, 32));
      mx = fmaxf(mx, __shfl_xor(mx, 8, 32));
      const float mnew = fmaxf(mrow[v], mx);
      const float sc = expf(mrow[v] - mnew);
      const float p0 = expf(a0 - mnew);
      const float p1 = expf(a1v - mnew);
      float rs = p0 + p1;
      rs += __shfl_xor(rs, 1, 32);
      rs += __shfl_xor(rs, 2, 32);
      rs += __shfl_xor(rs, 4, 32);
      rs += __shfl_xor(rs, 8, 32);
      lrow[v] = lrow[v] * sc + rs;
      mrow[v] = mnew;
#pragma unroll
      for (int j = 0; j < 4; ++j) acc[j][v] *= sc;
      pT[(h8 + v) * 32 + nlo]      = f2bf(p0);
      pT[(h8 + v) * 32 + 16 + nlo] = f2bf(p1);
    }
    // per-wave LDS is processed in order: ds_reads below see the writes above
    bf16x16 pa = load_frag_a(pT, 32, lane);   // C layout -> A layout transpose
#pragma unroll
    for (int j = 0; j < 4; ++j)
      acc[j] = wmma_bf16(pa, bv[j], acc[j]);
  }

  // normalize + write attn_out (b, s, NH*HD) bf16
#pragma unroll
  for (int j = 0; j < 4; ++j) {
    const int d = j * 16 + nlo;
#pragma unroll
    for (int v = 0; v < 8; ++v) {
      const int s = q0 + h8 + v;
      AO[((size_t)b * SS + s) * (NHH * HDD) + h * HDD + d] = f2bf(acc[j][v] / lrow[v]);
    }
  }
}

// ---- Stage 3: output projection  out = AO(4096x1024) @ Wo -> f32 ----------
__global__ __launch_bounds__(256) void out_gemm_kernel(const unsigned short* __restrict__ AO,
                                                       const unsigned short* __restrict__ WoT,
                                                       float* __restrict__ out) {
  const int lane = threadIdx.x & 31;
  const int wid  = blockIdx.x * 8 + (threadIdx.x >> 5);
  const int row0 = (wid & 127) * 32;
  const int col0 = (wid >> 7) * 64;
  const int KD = NHH * HDD;                   // 1024

  auto loadA = [&](bf16x16* a, int kk) {
#pragma unroll
    for (int i = 0; i < 2; ++i)
      a[i] = load_frag_a(AO + (size_t)(row0 + i * 16) * KD + kk, KD, lane);
  };
  auto loadB = [&](bf16x16* bw, int kk) {
#pragma unroll
    for (int j = 0; j < 4; ++j)
      bw[j] = load_frag_bt(WoT + (size_t)(col0 + j * 16) * KD + kk, KD, lane);
  };

  f32x8 acc[2][4] = {};
  bf16x16 aE[2], bE[4], aO[2], bO[4];
  loadA(aE, 0);
  loadB(bE, 0);
  loadA(aO, 32);
  loadB(bO, 32);
  for (int kk = 0; kk + 128 <= KD; kk += 64) {
#pragma unroll
    for (int i = 0; i < 2; ++i)
#pragma unroll
      for (int j = 0; j < 4; ++j)
        acc[i][j] = wmma_bf16(aE[i], bE[j], acc[i][j]);
    loadA(aE, kk + 64);
    loadB(bE, kk + 64);
#pragma unroll
    for (int i = 0; i < 2; ++i)
#pragma unroll
      for (int j = 0; j < 4; ++j)
        acc[i][j] = wmma_bf16(aO[i], bO[j], acc[i][j]);
    loadA(aO, kk + 96);
    loadB(bO, kk + 96);
  }
#pragma unroll
  for (int i = 0; i < 2; ++i)
#pragma unroll
    for (int j = 0; j < 4; ++j)
      acc[i][j] = wmma_bf16(aE[i], bE[j], acc[i][j]);
#pragma unroll
  for (int i = 0; i < 2; ++i)
#pragma unroll
    for (int j = 0; j < 4; ++j)
      acc[i][j] = wmma_bf16(aO[i], bO[j], acc[i][j]);

  const int m0 = (lane >> 4) * 8;
#pragma unroll
  for (int i = 0; i < 2; ++i)
#pragma unroll
    for (int j = 0; j < 4; ++j) {
      const int col = col0 + j * 16 + (lane & 15);
#pragma unroll
      for (int v = 0; v < 8; ++v)
        out[(size_t)(row0 + i * 16 + m0 + v) * HH + col] = acc[i][j][v];
    }
}

// ---- Host-side orchestration ----------------------------------------------
extern "C" void kernel_launch(void* const* d_in, const int* in_sizes, int n_in,
                              void* d_out, int out_size, void* d_ws, size_t ws_size,
                              hipStream_t stream) {
  (void)in_sizes; (void)n_in; (void)out_size; (void)ws_size;
  const float* X  = (const float*)d_in[0];
  // d_in[1] = attention_mask: pure causal, applied analytically in-kernel
  const float* Wq = (const float*)d_in[2];
  const float* Wk = (const float*)d_in[3];
  const float* Wv = (const float*)d_in[4];
  const float* Wo = (const float*)d_in[5];
  const float* lp = (const float*)d_in[6];
  float* out = (float*)d_out;

  char* base = (char*)d_ws;
  size_t off = 0;
  auto carve = [&](size_t elts) -> unsigned short* {
    unsigned short* p = (unsigned short*)(base + off);
    off = (off + elts * sizeof(unsigned short) + 255) & ~(size_t)255;
    return p;
  };
  unsigned short* Xbf = carve((size_t)4096 * 2048);        // X bf16
  unsigned short* WqT = carve((size_t)2048 * 2048);        // Wq^T
  unsigned short* WkT = carve((size_t)2048 * 2048);        // Wk^T
  unsigned short* WvT = carve((size_t)1024 * 2048);        // Wv^T
  unsigned short* WoT = carve((size_t)2048 * 1024);        // Wo^T
  unsigned short* Qb  = carve((size_t)2 * 32 * 2048 * 64); // (b, head32, s, d)
  unsigned short* Kb  = carve((size_t)2 * 32 * 2048 * 64);
  unsigned short* Vtb = carve((size_t)2 * 16 * 64 * 2048); // (b, head16, d, s)
  unsigned short* AO  = carve((size_t)4096 * 1024);        // attn_out bf16

  cast_bf16_kernel<<<dim3((4096 * 2048) / 256), dim3(256), 0, stream>>>(X, Xbf, 4096 * 2048);
  cast_transpose_kernel<<<dim3(64, 64), dim3(32, 8), 0, stream>>>(Wq, WqT, 2048, 2048);
  cast_transpose_kernel<<<dim3(64, 64), dim3(32, 8), 0, stream>>>(Wk, WkT, 2048, 2048);
  cast_transpose_kernel<<<dim3(32, 64), dim3(32, 8), 0, stream>>>(Wv, WvT, 2048, 1024);
  cast_transpose_kernel<<<dim3(64, 32), dim3(32, 8), 0, stream>>>(Wo, WoT, 1024, 2048);

  // waves = 128 row-tiles * (N/64) col-tiles; 8 waves per 256-thread block
  proj_gemm_kernel<<<dim3(512), dim3(256), 0, stream>>>(Xbf, WqT, Qb, 2048, 0);
  proj_gemm_kernel<<<dim3(512), dim3(256), 0, stream>>>(Xbf, WkT, Kb, 2048, 0);
  proj_gemm_kernel<<<dim3(256), dim3(256), 0, stream>>>(Xbf, WvT, Vtb, 1024, 1);

  diff_attn_kernel<<<dim3(SS / 16 / 4, NHH, BB), dim3(128), 0, stream>>>(Qb, Kb, Vtb, lp, AO);

  out_gemm_kernel<<<dim3(512), dim3(256), 0, stream>>>(AO, WoT, out);
}
